// LSTM_unit_6365141533233
// MI455X (gfx1250) — matrix operand order repaired and checked by reading, not compiled, not run on hardware
//
#include <hip/hip_runtime.h>
#include <stdint.h>

// ---------------- Types ----------------
typedef __bf16 bf16_t;
typedef bf16_t v16bf __attribute__((ext_vector_type(16)));
typedef float  v8f   __attribute__((ext_vector_type(8)));
typedef unsigned int u32x4 __attribute__((ext_vector_type(4)));

// ---------------- Problem dims ----------------
#define SIZE   2048
#define BATCH  4096
#define KDIM   4096   // packed K = [x | h] contribution
#define NGATE  8192   // 4 gates * 2048

// ---------------- GEMM tiling ----------------
#define BM 128            // batch rows per block
#define BN 32             // gate-columns per block (x4 gates)
#define BK 32             // K step == WMMA K
#define LDT 80u           // padded LDS row stride in bytes (64 data + 16 pad)
#define ATILE_BYTES (BM * LDT)        // 10240
#define BTILE_BYTES (128u * LDT)      // 4 gates * 32 rows -> 10240
#define BUF_BYTES   (ATILE_BYTES + BTILE_BYTES)
#define KTILES      (KDIM / BK)       // 128

// ---------------- Helpers ----------------
__device__ __forceinline__ unsigned short f2bf(float f) {
  unsigned u = __float_as_uint(f);
  u += 0x7fffu + ((u >> 16) & 1u);   // round-to-nearest-even
  return (unsigned short)(u >> 16);
}

__device__ __forceinline__ float fast_sigmoid(float x) {
  return 1.0f / (1.0f + __expf(-x));
}
__device__ __forceinline__ float fast_tanh(float x) {
  x = fminf(fmaxf(x, -15.0f), 15.0f);
  float e = __expf(2.0f * x);
  return (e - 1.0f) / (e + 1.0f);
}

// CDNA5 async global->LDS copy (per-lane 16B), tracked by ASYNCcnt.
__device__ __forceinline__ void async_b128(unsigned lds_off, const void* gptr) {
  asm volatile("global_load_async_to_lds_b128 %0, %1, off"
               :: "v"(lds_off), "v"(gptr) : "memory");
}
__device__ __forceinline__ void wait_async0() {
  asm volatile("s_wait_asynccnt 0" ::: "memory");
}

// ---------------- Kernel 1: fp32 -> bf16 pack  dst(row,4096) = [s0 row | s1 row]
__global__ void pack2_bf16_kernel(const float* __restrict__ s0,
                                  const float* __restrict__ s1,
                                  unsigned short* __restrict__ dst) {
  unsigned t   = blockIdx.x * blockDim.x + threadIdx.x;
  unsigned r   = t >> 10;                 // 1024 quads per 4096-wide row
  unsigned col = (t & 1023u) << 2;
  const float* s = (col < 2048u) ? (s0 + (size_t)r * 2048 + col)
                                 : (s1 + (size_t)r * 2048 + (col - 2048u));
  float4 f = *(const float4*)s;
  unsigned lo = (unsigned)f2bf(f.x) | ((unsigned)f2bf(f.y) << 16);
  unsigned hi = (unsigned)f2bf(f.z) | ((unsigned)f2bf(f.w) << 16);
  *(uint2*)(dst + (size_t)r * 4096 + col) = make_uint2(lo, hi);
}

// ---------------- Kernel 2: fused bf16 WMMA GEMM + LSTM epilogue ----------------
__global__ __launch_bounds__(256)
void lstm_gemm_kernel(const unsigned short* __restrict__ AB,   // 4096 x 4096 bf16
                      const unsigned short* __restrict__ Wc,   // 8192 x 4096 bf16
                      const float* __restrict__ bx,            // 4*2048
                      const float* __restrict__ bh,            // 4*2048
                      const float* __restrict__ cin,           // 4096 x 2048
                      float* __restrict__ outh,
                      float* __restrict__ outc) {
  __shared__ __align__(16) unsigned char smem[2 * BUF_BYTES];

  const unsigned tid  = threadIdx.x;
  const unsigned lane = tid & 31u;
  const unsigned wid  = tid >> 5;          // 8 waves
  const int m0 = blockIdx.x * BM;
  const int n0 = blockIdx.y * BN;
  const unsigned lds0 = (unsigned)(uintptr_t)(void*)smem;  // LDS byte offset of smem

  // Flat accumulators: index = g*2 + nt  (g = gate, nt = 16-col subtile)
  v8f acc[8];
#pragma unroll
  for (int i = 0; i < 8; ++i) acc[i] = (v8f)0.0f;

  // Hoisted per-lane fragment offsets (byte offsets within a buffer).
  const unsigned l15 = lane & 15u;
  const unsigned hiL = (lane >> 4) & 1u;  // 0: lanes 0-15, 1: lanes 16-31
  // A frag: wave's 16 rows; lanes0-15 K0..7|K16..23, lanes16-31 K8..15|K24..31
  const unsigned aOff = (wid * 16u + l15) * LDT + hiL * 16u;
  // B frag row base: lane n reads W row n, lanes0-15 K0..7|K16..23, lanes16-31 K8..15|K24..31
  const unsigned bOffBase = ATILE_BYTES + l15 * LDT + hiL * 16u;

  // Issue one K-tile (A: 128x32 bf16, B: 4 gates x 32x32 bf16) via async DMA.
  auto issue = [&](int kt, unsigned bufbase) {
    const int k0 = kt * BK;
#pragma unroll
    for (int i = 0; i < 2; ++i) {                 // A tile: 512 x 16B chunks
      unsigned c   = tid * 2u + (unsigned)i;
      unsigned row = c >> 2, off = (c & 3u) * 16u;
      async_b128(lds0 + bufbase + row * LDT + off,
                 (const unsigned char*)(AB + (size_t)(m0 + (int)row) * KDIM + k0) + off);
    }
#pragma unroll
    for (int i = 0; i < 2; ++i) {                 // B tile: 512 x 16B chunks
      unsigned c    = tid * 2u + (unsigned)i;
      unsigned row  = c >> 2, off = (c & 3u) * 16u;
      unsigned gate = row >> 5;
      unsigned wrow = gate * 2048u + (unsigned)n0 + (row & 31u);
      async_b128(lds0 + bufbase + ATILE_BYTES + row * LDT + off,
                 (const unsigned char*)(Wc + (size_t)wrow * KDIM + k0) + off);
    }
  };

  // Consume one K-tile. Preload A + ALL 8 B-frags into distinct registers
  // first, then issue the 8 WMMAs back-to-back so the matrix pipe is not
  // serialized on per-fragment s_wait_dscnt.
  auto compute = [&](unsigned bufbase) {
    const unsigned char* base = smem + bufbase;

    union { u32x4 q[2]; v16bf v; } af;
    {
      const unsigned char* p = base + aOff;
      af.q[0] = *(const u32x4*)p;
      af.q[1] = *(const u32x4*)(p + 32);
    }

    union { u32x4 q[2]; v16bf v; } bfr[8];
#pragma unroll
    for (int i = 0; i < 8; ++i) {
      const int g = i >> 1, nt = i & 1;
      const unsigned char* p =
          base + bOffBase + ((unsigned)g * 32u + (unsigned)nt * 16u) * LDT;
      bfr[i].q[0] = *(const u32x4*)p;
      bfr[i].q[1] = *(const u32x4*)(p + 32);
    }

#pragma unroll
    for (int i = 0; i < 8; ++i) {
      acc[i] = __builtin_amdgcn_wmma_f32_16x16x32_bf16(
          false, af.v, false, bfr[i].v, (short)0, acc[i], false, false);
    }
  };

  // ---- main loop: double-buffered async pipeline ----
  issue(0, 0u);
  for (int kt = 0; kt < KTILES; ++kt) {
    wait_async0();        // my async writes for tile kt are in LDS
    __syncthreads();      // everyone's writes visible; prev reads done
    if (kt + 1 < KTILES) issue(kt + 1, (unsigned)((kt + 1) & 1) * BUF_BYTES);
    compute((unsigned)(kt & 1) * BUF_BYTES);
  }

  // ---- epilogue: biases + LSTM nonlinearity, all 4 gates in-register ----
  // C/D layout: VGPR r -> M = r (lanes 0-15) / r+8 (lanes 16-31), N = lane&15
  const int mbase = m0 + (int)wid * 16 + ((lane < 16u) ? 0 : 8);
  const int nb    = n0 + (int)l15;
#pragma unroll
  for (int nt = 0; nt < 2; ++nt) {
    const int n = nb + nt * 16;
    const float b_i = bx[n]          + bh[n];
    const float b_f = bx[2048 + n]   + bh[2048 + n];
    const float b_o = bx[4096 + n]   + bh[4096 + n];
    const float b_c = bx[6144 + n]   + bh[6144 + n];
#pragma unroll
    for (int r = 0; r < 8; ++r) {
      const int m = mbase + r;
      float iv = fast_sigmoid(acc[0 * 2 + nt][r] + b_i);
      float fv = fast_sigmoid(acc[1 * 2 + nt][r] + b_f);
      float ov = fast_sigmoid(acc[2 * 2 + nt][r] + b_o);
      float cc = fast_tanh   (acc[3 * 2 + nt][r] + b_c);
      size_t idx = (size_t)m * SIZE + (size_t)n;
      float cn = fv * cin[idx] + iv * cc;
      outc[idx] = cn;
      outh[idx] = ov * fast_tanh(cn);
    }
  }
}

// ---------------- Launch ----------------
extern "C" void kernel_launch(void* const* d_in, const int* in_sizes, int n_in,
                              void* d_out, int out_size, void* d_ws, size_t ws_size,
                              hipStream_t stream) {
  (void)in_sizes; (void)n_in; (void)out_size; (void)ws_size;
  const float* x  = (const float*)d_in[0];
  const float* h  = (const float*)d_in[1];
  const float* c  = (const float*)d_in[2];
  const float* Wx = (const float*)d_in[3];   // (4,2048,2048) == (8192,2048) row-major
  const float* bx = (const float*)d_in[4];
  const float* Wh = (const float*)d_in[5];
  const float* bh = (const float*)d_in[6];

  float* outh = (float*)d_out;
  float* outc = outh + (size_t)BATCH * SIZE;

  unsigned short* AB = (unsigned short*)d_ws;                 // 4096x4096 bf16 (32 MB)
  unsigned short* Wc = AB + (size_t)BATCH * KDIM;             // 8192x4096 bf16 (64 MB)

  // Pack activations [x|h] and weights [Wx|Wh] to bf16.
  pack2_bf16_kernel<<<(BATCH * 1024) / 256, 256, 0, stream>>>(x, h, AB);
  pack2_bf16_kernel<<<(NGATE * 1024) / 256, 256, 0, stream>>>(Wx, Wh, Wc);

  dim3 grid(BATCH / BM, SIZE / BN);   // 32 x 64 blocks
  lstm_gemm_kernel<<<grid, 256, 0, stream>>>(AB, Wc, bx, bh, c, outh, outc);
}
